// PointTransformerBlock_35656818492088
// MI455X (gfx1250) — matrix operand order repaired and checked by reading, not compiled, not run on hardware
//
#include <hip/hip_runtime.h>
#include <hip/hip_bf16.h>

#define BB   4
#define NP   4096
#define KNN  16
#define DM   128
#define PTS  16                  // points per attention block
#define ROWS (PTS * KNN)         // 256 rows (point-major, 16 neighbors each)
#define TILES (PTS * (DM / 16))  // 16 M-tiles x 8 N-tiles = 128
#define NWAVES 8

typedef __attribute__((ext_vector_type(16))) _Float16 v16h;
typedef __attribute__((ext_vector_type(8)))  _Float16 v8h;
typedef __attribute__((ext_vector_type(8)))  float    v8f;

// ---- WMMA fragment loaders (layouts per CDNA5 ISA 7.12.2, wave32) ----------
// A 16x32 f16: lanes 0-15 row M=lane hold K 0..7 (v0-3) & 16..23 (v4-7);
//              lanes 16-31 same rows hold K 8..15 & 24..31.
__device__ __forceinline__ v16h frag_a_f32(const float* A, int m0, int k0, int lane) {
  v16h a;
  const float* p = A + (size_t)(m0 + (lane & 15)) * DM + k0 + ((lane >> 4) & 1) * 8;
#pragma unroll
  for (int j = 0; j < 8; ++j) {
    int kk = ((j < 4) ? 0 : 16) + 2 * (j & 3);
    a[2 * j]     = (_Float16)p[kk];
    a[2 * j + 1] = (_Float16)p[kk + 1];
  }
  return a;
}
__device__ __forceinline__ v16h frag_a_f16(const _Float16* A, int m0, int k0, int lane) {
  v16h a;
  const _Float16* p = A + (size_t)(m0 + (lane & 15)) * DM + k0 + ((lane >> 4) & 1) * 8;
#pragma unroll
  for (int j = 0; j < 8; ++j) {
    int kk = ((j < 4) ? 0 : 16) + 2 * (j & 3);
    a[2 * j]     = p[kk];
    a[2 * j + 1] = p[kk + 1];
  }
  return a;
}
// B 32x16 f16 (row-major K x N source): lanes 0-15 col N=lane hold K 0..7 & 16..23,
// lanes 16-31 hold K 8..15 & 24..31; 2 consecutive K per VGPR.
__device__ __forceinline__ v16h frag_b_f16(const _Float16* Bw, int k0, int n0, int lane) {
  v16h b;
  const _Float16* p = Bw + (size_t)(k0 + ((lane >> 4) & 1) * 8) * DM + n0 + (lane & 15);
#pragma unroll
  for (int j = 0; j < 8; ++j) {
    int kk = ((j < 4) ? 0 : 16) + 2 * (j & 3);
    b[2 * j]     = p[kk * DM];
    b[2 * j + 1] = p[(kk + 1) * DM];
  }
  return b;
}

// ---- f32 -> f16 weight conversion ------------------------------------------
__global__ void cvt_f16_kernel(const float* __restrict__ s, _Float16* __restrict__ d, int n) {
  const int i = blockIdx.x * blockDim.x + threadIdx.x;
  if (i < n) d[i] = (_Float16)s[i];
}

// ---- generic 128-K WMMA GEMM: Out[M,128] = A[M,128] @ Bw[128,128] (+bias,+resid)
// Each wave: one N-tile x two M-subtiles -> two independent WMMA chains
// (fills the D->A RAW hazard slots, shares the B fragment).
template <bool A_HALF, bool OUT_HALF>
__global__ __launch_bounds__(256) void gemm128_kernel(const void* __restrict__ Aptr,
                                                      const _Float16* __restrict__ Bw,
                                                      const float* __restrict__ bias,
                                                      const float* __restrict__ resid,
                                                      void* __restrict__ Outp) {
  const int lane = threadIdx.x & 31;
  const int wave = threadIdx.x >> 5;
  const int m0 = blockIdx.x * 32;
  const int n0 = wave * 16;
  const int col = n0 + (lane & 15);
  const int rowoff = ((lane >> 4) & 1) * 8;

  v8f acc0, acc1;
  const float bv = bias ? bias[col] : 0.0f;
#pragma unroll
  for (int j = 0; j < 8; ++j) { acc0[j] = bv; acc1[j] = bv; }

#pragma unroll
  for (int ks = 0; ks < DM; ks += 32) {
    v16h a0 = A_HALF ? frag_a_f16((const _Float16*)Aptr, m0, ks, lane)
                     : frag_a_f32((const float*)Aptr, m0, ks, lane);
    v16h a1 = A_HALF ? frag_a_f16((const _Float16*)Aptr, m0 + 16, ks, lane)
                     : frag_a_f32((const float*)Aptr, m0 + 16, ks, lane);
    v16h b = frag_b_f16(Bw, ks, n0, lane);
    acc0 = __builtin_amdgcn_wmma_f32_16x16x32_f16(false, a0, false, b, (short)0, acc0,
                                                  false, false);
    acc1 = __builtin_amdgcn_wmma_f32_16x16x32_f16(false, a1, false, b, (short)0, acc1,
                                                  false, false);
  }
#pragma unroll
  for (int j = 0; j < 8; ++j) {
    float v0 = acc0[j], v1 = acc1[j];
    const size_t o0 = (size_t)(m0 + rowoff + j) * DM + col;
    const size_t o1 = (size_t)(m0 + 16 + rowoff + j) * DM + col;
    if (resid) { v0 += resid[o0]; v1 += resid[o1]; }
    if (OUT_HALF) {
      ((_Float16*)Outp)[o0] = (_Float16)v0;
      ((_Float16*)Outp)[o1] = (_Float16)v1;
    } else {
      ((float*)Outp)[o0] = v0;
      ((float*)Outp)[o1] = v1;
    }
  }
}

// ---- brute-force KNN (top-16 by squared distance), LDS-tiled refs ----------
__global__ __launch_bounds__(256) void knn_kernel(const float* __restrict__ coord,
                                                  int* __restrict__ nidx) {
  __shared__ float sc[256 * 3];
  const int b = blockIdx.x / (NP / 256);
  const int n = (blockIdx.x % (NP / 256)) * 256 + threadIdx.x;
  const size_t gq = (size_t)b * NP + n;
  const float qx = coord[gq * 3 + 0], qy = coord[gq * 3 + 1], qz = coord[gq * 3 + 2];

  float best[KNN]; int bidx[KNN];
#pragma unroll
  for (int i = 0; i < KNN; ++i) { best[i] = 3.4e38f; bidx[i] = 0; }

  for (int t0 = 0; t0 < NP; t0 += 256) {
    const size_t gr = (size_t)b * NP + t0 + threadIdx.x;
    sc[threadIdx.x * 3 + 0] = coord[gr * 3 + 0];
    sc[threadIdx.x * 3 + 1] = coord[gr * 3 + 1];
    sc[threadIdx.x * 3 + 2] = coord[gr * 3 + 2];
    if (t0 + 256 < NP)  // pull next 3KB tile toward WGP while we scan this one
      __builtin_prefetch(&coord[(gr + 256) * 3], 0, 3);
    __syncthreads();
    for (int t = 0; t < 256; ++t) {
      const float dx = qx - sc[t * 3 + 0], dy = qy - sc[t * 3 + 1], dz = qz - sc[t * 3 + 2];
      float d2 = dx * dx + dy * dy + dz * dz;
      if (d2 < best[KNN - 1]) {
        int mi = t0 + t;
#pragma unroll
        for (int i = 0; i < KNN; ++i) {
          if (d2 < best[i]) {
            float tb = best[i]; best[i] = d2; d2 = tb;
            int   ti = bidx[i]; bidx[i] = mi; mi = ti;
          }
        }
      }
    }
    __syncthreads();
  }
#pragma unroll
  for (int i = 0; i < KNN; ++i) nidx[gq * KNN + i] = bidx[i];
}

// ---- one MLP layer on the 256x128 LDS tile: Dst = act(Asrc @ Bw + bias) ----
// Pairs adjacent N-tiles: shared A fragment, two independent WMMA chains.
__device__ __forceinline__ void mlp_layer_pair(const _Float16* Asrc, const _Float16* Bw,
                                               const float* bias, _Float16* Dst,
                                               bool relu, int wave, int lane) {
  for (int tp = wave; tp < TILES / 2; tp += NWAVES) {
    const int mt = (tp >> 2) * 16;
    const int nt0 = (tp & 3) * 32;
    const int c0 = nt0 + (lane & 15);
    v8f acc0, acc1;
    const float bv0 = bias[c0], bv1 = bias[c0 + 16];
#pragma unroll
    for (int j = 0; j < 8; ++j) { acc0[j] = bv0; acc1[j] = bv1; }
#pragma unroll
    for (int ks = 0; ks < DM; ks += 32) {
      v16h a  = frag_a_f16(Asrc, mt, ks, lane);
      v16h b0 = frag_b_f16(Bw, ks, nt0, lane);
      v16h b1 = frag_b_f16(Bw, ks, nt0 + 16, lane);
      acc0 = __builtin_amdgcn_wmma_f32_16x16x32_f16(false, a, false, b0, (short)0, acc0,
                                                    false, false);
      acc1 = __builtin_amdgcn_wmma_f32_16x16x32_f16(false, a, false, b1, (short)0, acc1,
                                                    false, false);
    }
    const int rowoff = ((lane >> 4) & 1) * 8;
#pragma unroll
    for (int j = 0; j < 8; ++j) {
      float v0 = acc0[j], v1 = acc1[j];
      if (relu) { v0 = v0 > 0.f ? v0 : 0.f; v1 = v1 > 0.f ? v1 : 0.f; }
      Dst[(mt + rowoff + j) * DM + c0]      = (_Float16)v0;
      Dst[(mt + rowoff + j) * DM + c0 + 16] = (_Float16)v1;
    }
  }
}

// ---- fused neighbor pipeline: pos_enc MLP, gamma MLP, softmax, weighted sum
__global__ __launch_bounds__(256) void pt_attn_kernel(
    const float* __restrict__ coord, const int* __restrict__ nidx,
    const _Float16* __restrict__ qh, const _Float16* __restrict__ kh,
    const _Float16* __restrict__ vh,
    const float* __restrict__ Wt1, const float* __restrict__ bt1,
    const _Float16* __restrict__ Wt2h, const float* __restrict__ bt2,
    const _Float16* __restrict__ Wg1h, const float* __restrict__ bg1,
    const _Float16* __restrict__ Wg2h, const float* __restrict__ bg2,
    float* __restrict__ attn_out) {
  __shared__ _Float16 sY[ROWS * DM];   // 64 KB: theta hidden, then q-k+pos
  __shared__ _Float16 sH[ROWS * DM];   // 64 KB: gamma hidden
  __shared__ _Float16 sP[ROWS * DM];   // 64 KB: pos_enc (kept for v+pos)
  __shared__ int sIdx[ROWS];

  const int b = blockIdx.x / (NP / PTS);
  const int n0 = (blockIdx.x % (NP / PTS)) * PTS;
  const int lane = threadIdx.x & 31;
  const int wave = threadIdx.x >> 5;
  const size_t gbase = (size_t)b * NP;

  // stage 0: idx, coord deltas, theta layer-1 (3 -> 128, relu) per row
  for (int r = threadIdx.x; r < ROWS; r += 256) {
    const int p = r >> 4, j = r & 15;
    const int n = n0 + p;
    const int m = nidx[(gbase + n) * KNN + j];
    sIdx[r] = m;
    const float dx = coord[(gbase + n) * 3 + 0] - coord[(gbase + m) * 3 + 0];
    const float dy = coord[(gbase + n) * 3 + 1] - coord[(gbase + m) * 3 + 1];
    const float dz = coord[(gbase + n) * 3 + 2] - coord[(gbase + m) * 3 + 2];
    for (int f = 0; f < DM; ++f) {
      const float h = dx * Wt1[f] + dy * Wt1[DM + f] + dz * Wt1[2 * DM + f] + bt1[f];
      sY[r * DM + f] = (_Float16)(h > 0.f ? h : 0.f);
    }
  }
  __syncthreads();

  // stage 1 (WMMA): sP = sY @ Wt2 + bt2
  mlp_layer_pair(sY, Wt2h, bt2, sP, false, wave, lane);
  __syncthreads();

  // stage 2: Y = q - k + pos_enc (packed f16 vector math)
  for (int r = threadIdx.x; r < ROWS; r += 256) {
    const int p = r >> 4;
    const int n = n0 + p;
    const int m = sIdx[r];
    const _Float16* qr = qh + (gbase + n) * DM;
    const _Float16* kr = kh + (gbase + m) * DM;
#pragma unroll
    for (int f = 0; f < DM; f += 8) {
      v8h qv = *(const v8h*)(qr + f);
      v8h kv = *(const v8h*)(kr + f);
      v8h pv = *(const v8h*)(&sP[r * DM + f]);
      *(v8h*)(&sY[r * DM + f]) = qv - kv + pv;
    }
  }
  __syncthreads();

  // stage 3 (WMMA): sH = relu(sY @ Wg1 + bg1)
  mlp_layer_pair(sY, Wg1h, bg1, sH, true, wave, lane);
  __syncthreads();

  // stage 4 (WMMA + in-register softmax over M=16 neighbors + weighted sum)
  const float scale = 0.08838834764831845f;  // 128^-0.5
  for (int tp = wave; tp < TILES / 2; tp += NWAVES) {
    const int mt = tp >> 2;  // point index within tile
    const int nt0 = (tp & 3) * 32;
    const int c0 = nt0 + (lane & 15);
    v8f acc0, acc1;
    const float bv0 = bg2[c0], bv1 = bg2[c0 + 16];
#pragma unroll
    for (int j = 0; j < 8; ++j) { acc0[j] = bv0; acc1[j] = bv1; }
#pragma unroll
    for (int ks = 0; ks < DM; ks += 32) {
      v16h a  = frag_a_f16(sH, mt * 16, ks, lane);
      v16h b0 = frag_b_f16(Wg2h, ks, nt0, lane);
      v16h b1 = frag_b_f16(Wg2h, ks, nt0 + 16, lane);
      acc0 = __builtin_amdgcn_wmma_f32_16x16x32_f16(false, a, false, b0, (short)0, acc0,
                                                    false, false);
      acc1 = __builtin_amdgcn_wmma_f32_16x16x32_f16(false, a, false, b1, (short)0, acc1,
                                                    false, false);
    }
    const int rowoff = ((lane >> 4) & 1) * 8;
    const int rbase = mt * 16 + rowoff;

    auto finish = [&](v8f acc, int col) {
      float mx = -3.4e38f;
#pragma unroll
      for (int j = 0; j < 8; ++j) { acc[j] *= scale; mx = fmaxf(mx, acc[j]); }
      mx = fmaxf(mx, __shfl_xor(mx, 16, 32));  // combine rows 0-7 / 8-15
      float e[8], s = 0.f;
#pragma unroll
      for (int j = 0; j < 8; ++j) { e[j] = __expf(acc[j] - mx); s += e[j]; }
      s += __shfl_xor(s, 16, 32);
      const float inv = 1.f / s;
      float out = 0.f;
#pragma unroll
      for (int j = 0; j < 8; ++j) {
        const int r = rbase + j;
        const int m = sIdx[r];
        const float vv = (float)vh[(gbase + m) * DM + col] + (float)sP[r * DM + col];
        out += e[j] * inv * vv;
      }
      out += __shfl_xor(out, 16, 32);
      if (lane < 16) attn_out[(gbase + n0 + mt) * DM + col] = out;
    };
    finish(acc0, c0);
    finish(acc1, c0 + 16);
  }
}

__global__ void copy_coord_kernel(const float* __restrict__ c, float* __restrict__ o, int n) {
  const int i = blockIdx.x * blockDim.x + threadIdx.x;
  if (i < n) o[i] = c[i];
}

extern "C" void kernel_launch(void* const* d_in, const int* in_sizes, int n_in,
                              void* d_out, int out_size, void* d_ws, size_t ws_size,
                              hipStream_t stream) {
  const float* feature = (const float*)d_in[0];
  const float* coord   = (const float*)d_in[1];
  const float* W1 = (const float*)d_in[2];  const float* b1 = (const float*)d_in[3];
  const float* W2 = (const float*)d_in[4];  const float* b2 = (const float*)d_in[5];
  const float* Wg1 = (const float*)d_in[6]; const float* bg1 = (const float*)d_in[7];
  const float* Wg2 = (const float*)d_in[8]; const float* bg2 = (const float*)d_in[9];
  const float* Wt1 = (const float*)d_in[10]; const float* bt1 = (const float*)d_in[11];
  const float* Wt2 = (const float*)d_in[12]; const float* bt2 = (const float*)d_in[13];
  const float* Wq = (const float*)d_in[14];
  const float* Wk = (const float*)d_in[15];
  const float* Wv = (const float*)d_in[16];

  const size_t BN = (size_t)BB * NP;  // 16384
  const size_t WN = (size_t)DM * DM;  // 16384 weights per matrix
  char* ws = (char*)d_ws;
  size_t off = 0;
  int* nidx = (int*)(ws + off);         off += BN * KNN * sizeof(int);
  _Float16* xh = (_Float16*)(ws + off); off += BN * DM * sizeof(_Float16);
  _Float16* qh = (_Float16*)(ws + off); off += BN * DM * sizeof(_Float16);
  _Float16* kh = (_Float16*)(ws + off); off += BN * DM * sizeof(_Float16);
  _Float16* vh = (_Float16*)(ws + off); off += BN * DM * sizeof(_Float16);
  float* aout = (float*)(ws + off);     off += BN * DM * sizeof(float);
  _Float16* W1h  = (_Float16*)(ws + off); off += WN * sizeof(_Float16);
  _Float16* Wqh  = (_Float16*)(ws + off); off += WN * sizeof(_Float16);
  _Float16* Wkh  = (_Float16*)(ws + off); off += WN * sizeof(_Float16);
  _Float16* Wvh  = (_Float16*)(ws + off); off += WN * sizeof(_Float16);
  _Float16* Wt2h = (_Float16*)(ws + off); off += WN * sizeof(_Float16);
  _Float16* Wg1h = (_Float16*)(ws + off); off += WN * sizeof(_Float16);
  _Float16* Wg2h = (_Float16*)(ws + off); off += WN * sizeof(_Float16);
  _Float16* W2h  = (_Float16*)(ws + off); off += WN * sizeof(_Float16);
  (void)ws_size; (void)in_sizes; (void)n_in; (void)out_size;

  // 0) weight pre-conversion to f16 (done every call; tiny)
  const int wgrid = (int)(WN / 256);
  cvt_f16_kernel<<<wgrid, 256, 0, stream>>>(W1, W1h, (int)WN);
  cvt_f16_kernel<<<wgrid, 256, 0, stream>>>(Wq, Wqh, (int)WN);
  cvt_f16_kernel<<<wgrid, 256, 0, stream>>>(Wk, Wkh, (int)WN);
  cvt_f16_kernel<<<wgrid, 256, 0, stream>>>(Wv, Wvh, (int)WN);
  cvt_f16_kernel<<<wgrid, 256, 0, stream>>>(Wt2, Wt2h, (int)WN);
  cvt_f16_kernel<<<wgrid, 256, 0, stream>>>(Wg1, Wg1h, (int)WN);
  cvt_f16_kernel<<<wgrid, 256, 0, stream>>>(Wg2, Wg2h, (int)WN);
  cvt_f16_kernel<<<wgrid, 256, 0, stream>>>(W2, W2h, (int)WN);

  // 1) KNN indices
  knn_kernel<<<(int)(BN / 256), 256, 0, stream>>>(coord, nidx);
  // 2) x = feature @ W1 + b1  (f32 in -> f16 out)
  gemm128_kernel<false, true><<<(int)(BN / 32), 256, 0, stream>>>(feature, W1h, b1, nullptr, xh);
  // 3) q/k/v projections (f16 in -> f16 out, no bias)
  gemm128_kernel<true, true><<<(int)(BN / 32), 256, 0, stream>>>(xh, Wqh, nullptr, nullptr, qh);
  gemm128_kernel<true, true><<<(int)(BN / 32), 256, 0, stream>>>(xh, Wkh, nullptr, nullptr, kh);
  gemm128_kernel<true, true><<<(int)(BN / 32), 256, 0, stream>>>(xh, Wvh, nullptr, nullptr, vh);
  // 4) fused pos-enc / gamma / softmax / aggregate
  pt_attn_kernel<<<(int)(BN / PTS), 256, 0, stream>>>(coord, nidx, qh, kh, vh,
                                                      Wt1, bt1, Wt2h, bt2,
                                                      Wg1h, bg1, Wg2h, bg2, aout);
  // 5) out = aout @ W2 + b2 + feature  -> d_out[0 : B*N*128]
  gemm128_kernel<false, false><<<(int)(BN / 32), 256, 0, stream>>>(aout, W2h, b2, feature, (float*)d_out);
  // 6) tuple tail: coord passthrough
  const int ncc = (int)(BN * 3);
  copy_coord_kernel<<<(ncc + 255) / 256, 256, 0, stream>>>(coord, (float*)d_out + BN * (size_t)DM, ncc);
}
